// Net_34041910788666
// MI455X (gfx1250) — matrix operand order repaired and checked by reading, not compiled, not run on hardware
//
#include <hip/hip_runtime.h>
#include <cstdint>

#define N_NODES  50000
#define N_EDGES  800000
#define N_GRAPHS 512
#define NUM_CLASSES 10

typedef __attribute__((ext_vector_type(16))) _Float16 v16h;
typedef __attribute__((ext_vector_type(8)))  float    v8f;

union Frag16 { v16h v; uint4 q[2]; };   // 32 bytes: 16 halves == 2x b128

// ---------------- degree / normalization ----------------
__global__ void k_init_deg(float* __restrict__ deg) {
    int n = blockIdx.x * blockDim.x + threadIdx.x;
    if (n < N_NODES) deg[n] = 1.0f;                 // self-loop
}

__global__ void k_count_deg(const int* __restrict__ dst, float* __restrict__ deg) {
    int e = blockIdx.x * blockDim.x + threadIdx.x;
    if (e < N_EDGES) atomicAdd(&deg[dst[e]], 1.0f);
}

__global__ void k_finish_deg(const float* __restrict__ deg,
                             float* __restrict__ inv_sqrt,
                             float* __restrict__ inv_deg) {
    int n = blockIdx.x * blockDim.x + threadIdx.x;
    if (n < N_NODES) {
        float d = deg[n];
        inv_sqrt[n] = rsqrtf(d);
        inv_deg[n]  = 1.0f / d;
    }
}

__global__ void k_norm_edges(const int* __restrict__ src, const int* __restrict__ dst,
                             const float* __restrict__ inv_sqrt,
                             float* __restrict__ norm_e) {
    int e = blockIdx.x * blockDim.x + threadIdx.x;
    if (e < N_EDGES) norm_e[e] = inv_sqrt[src[e]] * inv_sqrt[dst[e]];
}

// ---------------- one-time f32 -> f16 conversions ----------------
// x (N x 5) -> Hh (N x 32), zero-padded K
__global__ void k_cvt_x(const float* __restrict__ x, _Float16* __restrict__ Hh) {
    int gid = blockIdx.x * blockDim.x + threadIdx.x;
    if (gid >= N_NODES * 32) return;
    int n = gid >> 5, k = gid & 31;
    Hh[gid] = (k < 5) ? (_Float16)x[n * 5 + k] : (_Float16)0.0f;
}

// W (FIN x FOUT) -> Wth (FOUT x K) transposed f16, K = FIN padded to mult of 32
template <int FIN, int K, int FOUT>
__global__ void k_cvt_w(const float* __restrict__ W, _Float16* __restrict__ Wth) {
    int gid = blockIdx.x * blockDim.x + threadIdx.x;
    if (gid >= FOUT * K) return;
    int col = gid / K, k = gid - col * K;
    Wth[gid] = (k < FIN) ? (_Float16)W[k * FOUT + col] : (_Float16)0.0f;
}

// ---------------- WMMA GEMM: T(f32) = Hh(N x K f16) @ Wth^T(FOUT x K f16) ----
// One wave32 per 16x32 output tile: one A fragment, two B fragments, two
// accumulators. Fragments loaded as contiguous b128 pairs per the ISA layout:
//   A: lane half selects K +0/+8; halves [0..7]=K..K+7, [8..15]=K+16..K+23
//   B: same as A: lane half selects K +0/+8; halves [0..7]=K..K+7, [8..15]=K+16..K+23
template <int K, int FOUT>
__global__ void k_gemm_wmma(const _Float16* __restrict__ Hh,
                            const _Float16* __restrict__ Wth,
                            float* __restrict__ T) {
    const int tiles_n = FOUT >> 5;
    const int tiles_m = N_NODES >> 4;               // 3125, exact
    int wave = (blockIdx.x * blockDim.x + threadIdx.x) >> 5;
    int lane = threadIdx.x & 31;
    if (wave >= tiles_m * tiles_n) return;          // uniform per-wave exit

    int tm = wave / tiles_n;
    int tn = wave - tm * tiles_n;
    int sub     = lane & 15;
    int half_hi = lane >> 4;                        // 0: lanes 0-15, 1: lanes 16-31
    int row  = tm * 16 + sub;
    int col0 = tn * 32 + sub;
    int col1 = col0 + 16;

    const _Float16* arow = Hh  + (long)row  * K + (half_hi << 3);   // +8 halves
    const _Float16* b0p  = Wth + (long)col0 * K + (half_hi << 3);   // +8 halves
    const _Float16* b1p  = Wth + (long)col1 * K + (half_hi << 3);

    v8f c0 = {}, c1 = {};
#pragma unroll
    for (int k0 = 0; k0 < K; k0 += 32) {
        Frag16 a, b0, b1;
        a.q[0]  = *(const uint4*)(arow + k0);         // K = k0+off .. +7
        a.q[1]  = *(const uint4*)(arow + k0 + 16);    // K = k0+16+off .. +23
        b0.q[0] = *(const uint4*)(b0p + k0);
        b0.q[1] = *(const uint4*)(b0p + k0 + 16);
        b1.q[0] = *(const uint4*)(b1p + k0);
        b1.q[1] = *(const uint4*)(b1p + k0 + 16);
        c0 = __builtin_amdgcn_wmma_f32_16x16x32_f16(false, a.v, false, b0.v,
                                                    (short)0, c0, false, false);
        c1 = __builtin_amdgcn_wmma_f32_16x16x32_f16(false, a.v, false, b1.v,
                                                    (short)0, c1, false, false);
    }
    // C/D layout: VGPR r -> M = r + 8*half_hi, N = lane&15
#pragma unroll
    for (int r = 0; r < 8; ++r) {
        int m = r + (half_hi << 3);
        float* trow = T + ((long)(tm * 16 + m)) * FOUT;
        trow[col0] = c0[r];
        trow[col1] = c1[r];
    }
}

// ---------------- agg init: AGG = T*inv_deg + bias ----------------
__global__ void k_init_agg(const float* __restrict__ T, const float* __restrict__ inv_deg,
                           const float* __restrict__ bias, float* __restrict__ AGG,
                           int F, long total) {
    long gid = (long)blockIdx.x * blockDim.x + threadIdx.x;
    if (gid >= total) return;
    int n = (int)(gid / F);
    int f = (int)(gid - (long)n * F);
    AGG[gid] = T[gid] * inv_deg[n] + bias[f];
}

// ---------------- edge scatter: AGG[dst] += norm_e * T[src] ----------------
__global__ void k_edge_agg(const float* __restrict__ T, const int* __restrict__ src,
                           const int* __restrict__ dst, const float* __restrict__ norm_e,
                           float* __restrict__ AGG, int F) {
    long gid = (long)blockIdx.x * blockDim.x + threadIdx.x;
    int chunks = F >> 2;                             // float4 chunks per edge
    long e = gid / chunks;
    if (e >= N_EDGES) return;
    int c = (int)(gid - e * chunks) << 2;
    int s = src[e], d = dst[e];
    float w = norm_e[e];
    float4 v = *(const float4*)(T + (long)s * F + c);
    float* o = AGG + (long)d * F + c;
    atomicAdd(o + 0, v.x * w);
    atomicAdd(o + 1, v.y * w);
    atomicAdd(o + 2, v.z * w);
    atomicAdd(o + 3, v.w * w);
}

// ---------------- relu, writing next layer's f16 activations ----------------
__global__ void k_relu_cvt(const float* __restrict__ AGG, _Float16* __restrict__ Hh,
                           long total) {
    long gid = (long)blockIdx.x * blockDim.x + threadIdx.x;
    if (gid < total) Hh[gid] = (_Float16)fmaxf(AGG[gid], 0.0f);
}

// ---------------- pooling ----------------
__global__ void k_zero(float* __restrict__ p, int n) {
    int i = blockIdx.x * blockDim.x + threadIdx.x;
    if (i < n) p[i] = 0.0f;
}

__global__ void k_pool(const _Float16* __restrict__ H, const int* __restrict__ batch,
                       float* __restrict__ pooled, float* __restrict__ counts) {
    long gid = (long)blockIdx.x * blockDim.x + threadIdx.x;
    const int chunks = 256 / 8;                      // 8 halves per thread
    long n = gid / chunks;
    if (n >= N_NODES) return;
    int c = (int)(gid - n * chunks) << 3;
    int g = batch[n];
    const _Float16* p = H + n * 256 + c;
    float* o = pooled + (long)g * 256 + c;
#pragma unroll
    for (int i = 0; i < 8; ++i) atomicAdd(o + i, (float)p[i]);
    if (c == 0) atomicAdd(&counts[g], 1.0f);
}

// ---------------- FC + log_softmax (512x10, trivial) ----------------
__global__ void k_fc_logsoftmax(const float* __restrict__ pooled, const float* __restrict__ counts,
                                const float* __restrict__ Wfc, const float* __restrict__ bfc,
                                float* __restrict__ out) {
    int g = blockIdx.x * blockDim.x + threadIdx.x;
    if (g >= N_GRAPHS) return;
    float invc = 1.0f / fmaxf(counts[g], 1.0f);
    float logits[NUM_CLASSES];
#pragma unroll
    for (int c = 0; c < NUM_CLASSES; ++c) logits[c] = bfc[c];
    for (int k = 0; k < 256; ++k) {
        float p = pooled[(long)g * 256 + k] * invc;
#pragma unroll
        for (int c = 0; c < NUM_CLASSES; ++c) logits[c] += p * Wfc[k * NUM_CLASSES + c];
    }
    float m = logits[0];
#pragma unroll
    for (int c = 1; c < NUM_CLASSES; ++c) m = fmaxf(m, logits[c]);
    float s = 0.0f;
#pragma unroll
    for (int c = 0; c < NUM_CLASSES; ++c) s += __expf(logits[c] - m);
    float lse = m + __logf(s);
#pragma unroll
    for (int c = 0; c < NUM_CLASSES; ++c) out[(long)g * NUM_CLASSES + c] = logits[c] - lse;
}

// ---------------- launch ----------------
static inline int cdiv_i(long a, int b) { return (int)((a + b - 1) / b); }

extern "C" void kernel_launch(void* const* d_in, const int* in_sizes, int n_in,
                              void* d_out, int out_size, void* d_ws, size_t ws_size,
                              hipStream_t stream) {
    const float* x     = (const float*)d_in[0];
    const int*   src   = (const int*)d_in[1];
    const int*   dst   = src + N_EDGES;             // edge_index row-major (2, E)
    const int*   batch = (const int*)d_in[2];
    const float* W[4]  = {(const float*)d_in[3], (const float*)d_in[5],
                          (const float*)d_in[7], (const float*)d_in[9]};
    const float* b[4]  = {(const float*)d_in[4], (const float*)d_in[6],
                          (const float*)d_in[8], (const float*)d_in[10]};
    const float* Wfc   = (const float*)d_in[11];
    const float* bfc   = (const float*)d_in[12];
    float*       out   = (float*)d_out;

    float* ws       = (float*)d_ws;
    float* deg      = ws;  ws += N_NODES;
    float* inv_sqrt = ws;  ws += N_NODES;
    float* inv_deg  = ws;  ws += N_NODES;
    float* norm_e   = ws;  ws += N_EDGES;
    float* pooled   = ws;  ws += N_GRAPHS * 256;
    float* counts   = ws;  ws += N_GRAPHS;
    ws = (float*)(((uintptr_t)ws + 255) & ~(uintptr_t)255);
    float* T   = ws;  ws += (size_t)N_NODES * 256;              // GEMM output (f32)
    float* AGG = ws;  ws += (size_t)N_NODES * 256;              // aggregation (f32)
    _Float16* HhA = (_Float16*)ws;  ws += (size_t)N_NODES * 128; // f16 ping (<=256 halves/node)
    _Float16* HhB = (_Float16*)ws;  ws += (size_t)N_NODES * 128; // f16 pong
    _Float16* Wth = (_Float16*)ws;  ws += (size_t)256 * 128 / 2; // f16 W^T, K-padded

    const int TB = 256;

    // degrees + per-edge norms (shared across layers)
    k_init_deg  <<<cdiv_i(N_NODES, TB), TB, 0, stream>>>(deg);
    k_count_deg <<<cdiv_i(N_EDGES, TB), TB, 0, stream>>>(dst, deg);
    k_finish_deg<<<cdiv_i(N_NODES, TB), TB, 0, stream>>>(deg, inv_sqrt, inv_deg);
    k_norm_edges<<<cdiv_i(N_EDGES, TB), TB, 0, stream>>>(src, dst, inv_sqrt, norm_e);

    // x -> f16, zero-padded K=32
    k_cvt_x<<<cdiv_i((long)N_NODES * 32, TB), TB, 0, stream>>>(x, HhA);

    const int KP[4]   = {32, 32, 64, 128};          // padded K per layer
    const int FOUT[4] = {32, 64, 128, 256};
    _Float16* hin  = HhA;
    _Float16* hout = HhB;

    for (int l = 0; l < 4; ++l) {
        int waves   = (N_NODES / 16) * (FOUT[l] / 32);
        int gblocks = (waves + 7) / 8;               // 8 waves (256 threads) / block
        int wels    = FOUT[l] * KP[l];
        switch (l) {
            case 0:
                k_cvt_w<5, 32, 32>    <<<cdiv_i(wels, TB), TB, 0, stream>>>(W[l], Wth);
                k_gemm_wmma<32, 32>   <<<gblocks, TB, 0, stream>>>(hin, Wth, T);
                break;
            case 1:
                k_cvt_w<32, 32, 64>   <<<cdiv_i(wels, TB), TB, 0, stream>>>(W[l], Wth);
                k_gemm_wmma<32, 64>   <<<gblocks, TB, 0, stream>>>(hin, Wth, T);
                break;
            case 2:
                k_cvt_w<64, 64, 128>  <<<cdiv_i(wels, TB), TB, 0, stream>>>(W[l], Wth);
                k_gemm_wmma<64, 128>  <<<gblocks, TB, 0, stream>>>(hin, Wth, T);
                break;
            case 3:
                k_cvt_w<128, 128, 256><<<cdiv_i(wels, TB), TB, 0, stream>>>(W[l], Wth);
                k_gemm_wmma<128, 256> <<<gblocks, TB, 0, stream>>>(hin, Wth, T);
                break;
        }
        long nf = (long)N_NODES * FOUT[l];
        k_init_agg<<<cdiv_i(nf, TB), TB, 0, stream>>>(T, inv_deg, b[l], AGG, FOUT[l], nf);
        long et = (long)N_EDGES * (FOUT[l] / 4);
        k_edge_agg<<<cdiv_i(et, TB), TB, 0, stream>>>(T, src, dst, norm_e, AGG, FOUT[l]);
        k_relu_cvt<<<cdiv_i(nf, TB), TB, 0, stream>>>(AGG, hout, nf);
        _Float16* tmp = hin; hin = hout; hout = tmp;
    }

    // global mean pool + FC + log_softmax (final activations: hin, f16, 256 wide)
    k_zero<<<cdiv_i(N_GRAPHS * 256 + N_GRAPHS, TB), TB, 0, stream>>>(pooled, N_GRAPHS * 256 + N_GRAPHS);
    long pt = (long)N_NODES * 32;
    k_pool<<<cdiv_i(pt, TB), TB, 0, stream>>>(hin, batch, pooled, counts);
    k_fc_logsoftmax<<<cdiv_i(N_GRAPHS, TB), TB, 0, stream>>>(pooled, counts, Wfc, bfc, out);
}
